// SelfAttention_6485400617388
// MI455X (gfx1250) — matrix operand order repaired and checked by reading, not compiled, not run on hardware
//
#include <hip/hip_runtime.h>
#include <cmath>

typedef unsigned short u16;
typedef __attribute__((ext_vector_type(16))) __bf16 v16bf;
typedef __attribute__((ext_vector_type(8)))  float  v8f;

union BFrag { v16bf v; uint4 q[2]; };
union CFrag { v8f v; float f[8]; };

__device__ __forceinline__ u16 f2bf(float f) {
  unsigned u = __float_as_uint(f);
  u += 0x7fffu + ((u >> 16) & 1u);          // round-to-nearest-even
  return (u16)(u >> 16);
}

__device__ __forceinline__ v8f wmma_bf16(const BFrag& a, const BFrag& b, v8f c) {
  // D = A(16x32 bf16) x B(32x16 bf16) + C(16x16 f32)
  return __builtin_amdgcn_wmma_f32_16x16x32_bf16(
      false, a.v, false, b.v, (short)0, c, false, false);
}

// LDS byte offset of a __shared__ location (group-segment address space).
__device__ __forceinline__ unsigned lds_off(void* p) {
  return (unsigned)(unsigned long long)(__attribute__((address_space(3))) void*)p;
}

// CDNA5 async DMA: 16B per lane, memory -> LDS, tracked by ASYNCcnt (in-order).
__device__ __forceinline__ void async_g2l_b128(unsigned ldsDst, const void* gSrc) {
  asm volatile("global_load_async_to_lds_b128 %0, %1, off"
               :: "v"(ldsDst), "v"((unsigned long long)(size_t)gSrc)
               : "memory");
}
template <int N>
__device__ __forceinline__ void wait_async() {
  asm volatile("s_wait_asynccnt %0" :: "i"(N) : "memory");
}

// ---------------------------------------------------------------- converts
__global__ __launch_bounds__(256) void cvt_bf16_kernel(
    const float* __restrict__ in, u16* __restrict__ out, int n4) {
  int i = blockIdx.x * 256 + threadIdx.x;
  int stride = gridDim.x * 256;
  for (; i < n4; i += stride) {
    float4 v = ((const float4*)in)[i];
    ushort4 o;
    o.x = f2bf(v.x); o.y = f2bf(v.y); o.z = f2bf(v.z); o.w = f2bf(v.w);
    ((ushort4*)out)[i] = o;
  }
}

// out[n][k] = bf16(in[k][n]); 32x32 LDS-tiled transpose, both sides coalesced.
__global__ __launch_bounds__(256) void cvt_transpose_kernel(
    const float* __restrict__ in, u16* __restrict__ out, int K, int N) {
  __shared__ u16 tile[32][33];
  const int k0 = blockIdx.y * 32, n0 = blockIdx.x * 32;
  const int c = threadIdx.x & 31, r8 = threadIdx.x >> 5;
  #pragma unroll
  for (int i = 0; i < 4; ++i) {
    int r = r8 + i * 8;
    tile[r][c] = f2bf(in[(size_t)(k0 + r) * N + n0 + c]);
  }
  __syncthreads();
  #pragma unroll
  for (int i = 0; i < 4; ++i) {
    int r = r8 + i * 8;
    out[(size_t)(n0 + r) * K + k0 + c] = tile[c][r];
  }
}

// Vt[b][e][s] = qkv[b][s][2H + e] (bf16 -> bf16), 32x32 tiled.
__global__ __launch_bounds__(256) void transpose_v_kernel(
    const u16* __restrict__ qkv, u16* __restrict__ Vt) {
  constexpr int S = 2048, H = 1024, H3 = 3 * H;
  __shared__ u16 tile[32][33];
  const int s0 = blockIdx.x * 32, e0 = blockIdx.y * 32, b = blockIdx.z;
  const int c = threadIdx.x & 31, r8 = threadIdx.x >> 5;
  #pragma unroll
  for (int i = 0; i < 4; ++i) {
    int r = r8 + i * 8;                       // s index
    tile[r][c] = qkv[(size_t)(b * S + s0 + r) * H3 + 2 * H + e0 + c];
  }
  __syncthreads();
  #pragma unroll
  for (int i = 0; i < 4; ++i) {
    int r = r8 + i * 8;                       // e index
    Vt[(size_t)(b * H + e0 + r) * S + s0 + c] = tile[c][r];
  }
}

// ---------------------------------------------------------------- GEMM
// C[M,N] = A[M,K](bf16) * Bt[N,K](bf16, pre-transposed) + bias[N].
// Block 256x128, BK=32, double-buffered LDS; all staging via async DMA.
// 8 waves in 4x2 layout; each wave owns a 64x64 output slab = 16 WMMA tiles.
__global__ __launch_bounds__(256, 1) void gemm_bf16_kernel(
    const u16* __restrict__ A, const u16* __restrict__ Bt,
    const float* __restrict__ bias, void* __restrict__ Cout,
    int M, int N, int K, int out_is_bf16)
{
  __shared__ u16 As [2][256][40];   // row stride 80B: 16B aligned
  __shared__ u16 Bts[2][128][40];   // [n][k], same layout

  const int tid  = threadIdx.x;
  const int lane = tid & 31;
  const int wave = tid >> 5;
  const int mb = blockIdx.y * 256;
  const int nb = blockIdx.x * 128;
  const int wrow = wave >> 1;        // 0..3 -> 64-row slab
  const int wcol = wave & 1;         // 0..1 -> 64-col slab
  const int ar = lane & 15;
  const int ak = (lane >> 4) * 8;
  const int bc = lane & 15;
  const int bk = (lane >> 4) * 8;
  const int rh = (lane >> 4) * 8;

  constexpr unsigned ABUF = 256 * 40 * 2;    // 20480 B
  constexpr unsigned BBUF = 128 * 40 * 2;    // 10240 B

  // staging: each thread moves 4x16B of A and 2x16B of B per k-step
  const int r0 = tid >> 2;                   // 0..63
  const int c0 = (tid & 3) * 8;
  const unsigned aDst = lds_off(&As [0][r0][c0]);
  const unsigned bDst = lds_off(&Bts[0][r0][c0]);
  const u16* aSrc = A  + (size_t)(mb + r0) * K + c0;
  const u16* bSrc = Bt + (size_t)(nb + r0) * K + c0;

  CFrag acc[4][4];
  #pragma unroll
  for (int mt = 0; mt < 4; ++mt)
    #pragma unroll
    for (int nt = 0; nt < 4; ++nt)
      #pragma unroll
      for (int j = 0; j < 8; ++j) acc[mt][nt].f[j] = 0.0f;

  auto stage = [&](int buf, int kb) {
    #pragma unroll
    for (int i = 0; i < 4; ++i)
      async_g2l_b128(aDst + buf * ABUF + i * 64 * 80,
                     aSrc + (size_t)i * 64 * K + kb);
    #pragma unroll
    for (int i = 0; i < 2; ++i)
      async_g2l_b128(bDst + buf * BBUF + i * 64 * 80,
                     bSrc + (size_t)i * 64 * K + kb);
    if (kb + 32 < K) {                        // prefetch tile after next into GL2
      __builtin_prefetch(aSrc + kb + 32, 0, 1);
      __builtin_prefetch(bSrc + kb + 32, 0, 1);
    }
  };

  stage(0, 0);                                 // prologue
  const int nk = K / 32;
  for (int t = 0; t < nk; ++t) {
    wait_async<0>();
    __syncthreads();
    if (t + 1 < nk) stage((t + 1) & 1, (t + 1) * 32);   // DMA overlaps compute

    const int buf = t & 1;
    BFrag af[4];
    #pragma unroll
    for (int mt = 0; mt < 4; ++mt) {
      const u16* p = &As[buf][wrow * 64 + mt * 16 + ar][0];
      af[mt].q[0] = *(const uint4*)(p + ak);
      af[mt].q[1] = *(const uint4*)(p + ak + 16);
    }
    #pragma unroll
    for (int nt = 0; nt < 4; ++nt) {          // stream B frags, reuse all A frags
      BFrag bf;
      const u16* p = &Bts[buf][wcol * 64 + nt * 16 + bc][0];
      bf.q[0] = *(const uint4*)(p + bk);
      bf.q[1] = *(const uint4*)(p + bk + 16);
      #pragma unroll
      for (int mt = 0; mt < 4; ++mt)
        acc[mt][nt].v = wmma_bf16(af[mt], bf, acc[mt][nt].v);
    }
  }

  #pragma unroll
  for (int mt = 0; mt < 4; ++mt) {
    #pragma unroll
    for (int nt = 0; nt < 4; ++nt) {
      int col = nb + wcol * 64 + nt * 16 + bc;
      float bv = bias ? bias[col] : 0.0f;
      #pragma unroll
      for (int j = 0; j < 8; ++j) {
        int row = mb + wrow * 64 + mt * 16 + rh + j;
        float val = acc[mt][nt].f[j] + bv;
        if (out_is_bf16) ((u16*)Cout)[(size_t)row * N + col] = f2bf(val);
        else             ((float*)Cout)[(size_t)row * N + col] = val;
      }
    }
  }
}

// ---------------------------------------------------------------- flash attention
// One WG = (batch b, 16-row q tile). 8 waves; wave w owns H-cols [w*128,+128).
// K staged per 512-wide e-half; V (pre-transposed Vt[b][e][s]) staged whole tile.
// ASYNCcnt completes in order -> partial wait lets score math overlap the V DMA.
__global__ __launch_bounds__(256, 1) void attn_kernel(
    const u16* __restrict__ qkv, const u16* __restrict__ Vt,
    u16* __restrict__ ctx, float scale)
{
  constexpr int TQ = 16, TK = 32, HD = 1024, SEQ = 2048, H3 = 3 * HD;
  __shared__ u16 Qs [TQ][HD + 8];     // row stride 2064B
  __shared__ u16 Ks [TK][512 + 8];    // row stride 1040B
  __shared__ u16 Vts[HD][TK + 8];     // [e][key], row stride 80B
  __shared__ u16 Ps [TQ][40];         // exp(scores) tile, bf16
  __shared__ float Ssc[TQ][TK];       // fp32 score reduction buffer
  __shared__ float mrow[TQ], lrow[TQ], arow[TQ];

  const int tid  = threadIdx.x;
  const int lane = tid & 31;
  const int wave = tid >> 5;
  const int b  = blockIdx.y;
  const int qt = blockIdx.x;
  const int ar = lane & 15;
  const int ak = (lane >> 4) * 8;
  const int bc = lane & 15;
  const int bk = (lane >> 4) * 8;
  const int rh = (lane >> 4) * 8;
  const int colBase = wave * 128;

  // ---- stage Q tile (16 x 1024) via async DMA
  const u16* qbase = qkv + (size_t)(b * SEQ + qt * TQ) * H3;
  {
    const int qr0 = tid >> 7;                // 0..1, +2 per step
    const int qc0 = (tid & 127) * 8;
    const unsigned qDst = lds_off(&Qs[qr0][qc0]);
    const u16* qSrc = qbase + (size_t)qr0 * H3 + qc0;
    #pragma unroll
    for (int it = 0; it < 8; ++it)
      async_g2l_b128(qDst + it * 2 * 2064, qSrc + (size_t)it * 2 * H3);
  }
  if (tid < TQ) { mrow[tid] = -3.0e38f; lrow[tid] = 0.0f; }

  CFrag acc[8];
  #pragma unroll
  for (int t = 0; t < 8; ++t)
    #pragma unroll
    for (int j = 0; j < 8; ++j) acc[t].f[j] = 0.0f;

  // staging maps
  const int kr0 = tid >> 6;                  // 0..3, +4 per step
  const int kc0 = (tid & 63) * 8;
  const unsigned kDst = lds_off(&Ks[kr0][kc0]);
  const u16* kPtr = qkv + (size_t)b * SEQ * H3 + HD;           // K base (advances)

  const int ve0 = tid >> 2;                  // 0..63, +64 per step
  const int vg  = (tid & 3) * 8;             // key subgroup (16B)
  const unsigned vDst = lds_off(&Vts[ve0][vg]);
  const u16* vSrc0 = Vt + (size_t)(b * HD + ve0) * SEQ + vg;   // + kt*TK

  auto issueK = [&](int e0h) {
    const u16* kSrc = kPtr + (size_t)kr0 * H3 + e0h + kc0;
    #pragma unroll
    for (int it = 0; it < 8; ++it)
      async_g2l_b128(kDst + it * 4 * 1040, kSrc + (size_t)it * 4 * H3);
  };

  auto scores = [&](int e0h) {
    CFrag sc[2];
    #pragma unroll
    for (int nt = 0; nt < 2; ++nt)
      #pragma unroll
      for (int j = 0; j < 8; ++j) sc[nt].f[j] = 0.0f;
    #pragma unroll
    for (int ec = 0; ec < 2; ++ec) {
      const int el = wave * 64 + ec * 32;
      BFrag a;
      const u16* qp = &Qs[ar][e0h + el];
      a.q[0] = *(const uint4*)(qp + ak);
      a.q[1] = *(const uint4*)(qp + ak + 16);
      #pragma unroll
      for (int nt = 0; nt < 2; ++nt) {
        BFrag kf;
        const u16* kp = &Ks[nt * 16 + bc][el];
        kf.q[0] = *(const uint4*)(kp + bk);
        kf.q[1] = *(const uint4*)(kp + bk + 16);
        sc[nt].v = wmma_bf16(a, kf, sc[nt].v);
      }
    }
    #pragma unroll
    for (int nt = 0; nt < 2; ++nt)
      #pragma unroll
      for (int j = 0; j < 8; ++j)
        atomicAdd(&Ssc[rh + j][nt * 16 + bc], sc[nt].f[j]);    // ds_add_f32
  };

  for (int kt = 0; kt < SEQ / TK; ++kt) {
    __syncthreads();                         // prev tile fully consumed
    issueK(0);                               // 8 asyncs: K half 0
    {                                        // 16 asyncs: whole V tile
      const u16* vSrc = vSrc0 + kt * TK;
      #pragma unroll
      for (int it = 0; it < 16; ++it)
        async_g2l_b128(vDst + it * 64 * 80, vSrc + (size_t)it * 64 * SEQ);
    }
    for (int i = tid; i < TQ * TK; i += 256) ((float*)Ssc)[i] = 0.0f;
    if (kt + 1 < SEQ / TK) {                 // prefetch next tile into GL2
      __builtin_prefetch(kPtr + (size_t)(TK + kr0) * H3 + kc0, 0, 1);
      __builtin_prefetch(vSrc0 + (kt + 1) * TK, 0, 1);
    }
    wait_async<16>();                        // K half 0 retired (in-order); V in flight
    __syncthreads();
    scores(0);
    __syncthreads();                         // all waves done reading K half 0
    issueK(512);                             // 8 asyncs: K half 1
    wait_async<0>();                         // V + K half 1 complete
    __syncthreads();
    scores(512);
    __syncthreads();

    // online softmax, one thread per q row (all in wave 0)
    if (tid < TQ) {
      const int r = tid;
      float m_old = mrow[r];
      float mx = m_old;
      #pragma unroll
      for (int j = 0; j < TK; ++j) mx = fmaxf(mx, Ssc[r][j] * scale);
      float alpha = __expf(m_old - mx);
      float sum = 0.0f;
      #pragma unroll
      for (int j = 0; j < TK; ++j) {
        float p = __expf(Ssc[r][j] * scale - mx);
        sum += p;
        Ps[r][j] = f2bf(p);
      }
      mrow[r] = mx;
      lrow[r] = lrow[r] * alpha + sum;
      arow[r] = alpha;
    }
    __syncthreads();

    // rescale accumulators and add P @ V
    float al[8];
    #pragma unroll
    for (int j = 0; j < 8; ++j) al[j] = arow[rh + j];
    BFrag pa;
    {
      const u16* pp = &Ps[ar][0];
      pa.q[0] = *(const uint4*)(pp + ak);
      pa.q[1] = *(const uint4*)(pp + ak + 16);
    }
    #pragma unroll
    for (int nt = 0; nt < 8; ++nt) {
      #pragma unroll
      for (int j = 0; j < 8; ++j) acc[nt].f[j] *= al[j];
      BFrag vf;
      const u16* vp = &Vts[colBase + nt * 16 + bc][0];
      vf.q[0] = *(const uint4*)(vp + bk);
      vf.q[1] = *(const uint4*)(vp + bk + 16);
      acc[nt].v = wmma_bf16(pa, vf, acc[nt].v);
    }

    kPtr += (size_t)TK * H3;
  }

  // finalize: divide by row sums, emit bf16 ctx
  float li[8];
  #pragma unroll
  for (int j = 0; j < 8; ++j) li[j] = 1.0f / lrow[rh + j];
  #pragma unroll
  for (int nt = 0; nt < 8; ++nt) {
    int col = colBase + nt * 16 + bc;
    #pragma unroll
    for (int j = 0; j < 8; ++j) {
      int row = qt * TQ + rh + j;
      ctx[(size_t)(b * SEQ + row) * HD + col] = f2bf(acc[nt].f[j] * li[j]);
    }
  }
}

// ---------------------------------------------------------------- launch
extern "C" void kernel_launch(void* const* d_in, const int* in_sizes, int n_in,
                              void* d_out, int out_size, void* d_ws, size_t ws_size,
                              hipStream_t stream) {
  constexpr int B = 4, S = 2048, D = 1024, H = 1024;
  const float* x  = (const float*)d_in[0];
  const float* W1 = (const float*)d_in[1];
  const float* b1 = (const float*)d_in[2];
  const float* W2 = (const float*)d_in[3];
  const float* b2 = (const float*)d_in[4];
  float* out = (float*)d_out;

  char* ws = (char*)d_ws;
  size_t off = 0;
  auto wsalloc = [&](size_t bytes) {
    void* p = ws + off;
    off = (off + bytes + 255) & ~(size_t)255;
    return p;
  };
  u16* xb   = (u16*)wsalloc((size_t)B * S * D * 2);       // 16 MB
  u16* w1t  = (u16*)wsalloc((size_t)D * 3 * H * 2);       //  6 MB  [3H][D]
  u16* w2t  = (u16*)wsalloc((size_t)H * H * 2);           //  2 MB  [H][H]
  u16* qkvb = (u16*)wsalloc((size_t)B * S * 3 * H * 2);   // 50 MB
  u16* ctxb = (u16*)wsalloc((size_t)B * S * H * 2);       // 17 MB
  u16* vtb  = (u16*)wsalloc((size_t)B * H * S * 2);       // 17 MB  [b][e][s]

  cvt_bf16_kernel<<<2048, 256, 0, stream>>>(x, xb, (B * S * D) / 4);
  cvt_transpose_kernel<<<dim3(3 * H / 32, D / 32), 256, 0, stream>>>(W1, w1t, D, 3 * H);
  cvt_transpose_kernel<<<dim3(H / 32, H / 32), 256, 0, stream>>>(W2, w2t, H, H);

  // qkv = x @ W1 + b1   (bf16 out)
  gemm_bf16_kernel<<<dim3(3 * H / 128, B * S / 256), 256, 0, stream>>>(
      xb, w1t, b1, qkvb, B * S, 3 * H, D, 1);

  // V -> Vt[b][e][s]
  transpose_v_kernel<<<dim3(S / 32, H / 32, B), 256, 0, stream>>>(qkvb, vtb);

  // fused flash attention -> ctx (bf16)
  const float scale = 1.0f / sqrtf(2.0f * (float)D);
  attn_kernel<<<dim3(S / 16, B), 256, 0, stream>>>(qkvb, vtb, ctxb, scale);

  // out = ctx @ W2 + b2  (f32 out)
  gemm_bf16_kernel<<<dim3(H / 128, B * S / 256), 256, 0, stream>>>(
      ctxb, w2t, b2, out, B * S, H, H, 0);
}
